// TransformerBlock_42889543418112
// MI455X (gfx1250) — matrix operand order repaired and checked by reading, not compiled, not run on hardware
//
#include <hip/hip_runtime.h>
#include <hip/hip_bf16.h>
#include <math.h>

// ---------------------------------------------------------------------------
// CDNA5 (gfx1250) transformer block: bf16 WMMA GEMMs (async-to-LDS, double
// buffered) + flash attention.
// ---------------------------------------------------------------------------

typedef __attribute__((ext_vector_type(16))) __bf16 v16bf;
typedef __attribute__((ext_vector_type(8)))  float  v8f;

union Frag16 { v16bf v; unsigned int u[8]; };
union FragC  { v8f  v; float        f[8]; };

__device__ __forceinline__ unsigned short f2bf(float x) {
  unsigned int u = __float_as_uint(x);
  u += 0x7FFFu + ((u >> 16) & 1u);          // round-to-nearest-even
  return (unsigned short)(u >> 16);
}

__device__ __forceinline__ float redmax16(float v) {
#pragma unroll
  for (int off = 1; off < 16; off <<= 1) v = fmaxf(v, __shfl_xor(v, off, 32));
  return v;
}
__device__ __forceinline__ float redsum16(float v) {
#pragma unroll
  for (int off = 1; off < 16; off <<= 1) v += __shfl_xor(v, off, 32);
  return v;
}

// Flat LDS pointer -> raw LDS byte offset (aperture: offset is the low 32 address bits).
__device__ __forceinline__ unsigned lds_lo(const void* p) {
  return (unsigned)(unsigned long long)p;
}
// Async global -> LDS 128-bit copy (per-lane), tracked by ASYNCcnt.
__device__ __forceinline__ void async_b128(unsigned lds, const void* g) {
  asm volatile("global_load_async_to_lds_b128 %0, %1, off"
               :: "v"(lds), "v"(g) : "memory");
}
__device__ __forceinline__ void wait_async0() {
  asm volatile("s_wait_asynccnt 0x0" ::: "memory");
}

// ---------------------------------------------------------------------------
// Weight cast + transpose: Wt[n][k] = bf16(W[k][n]).  Grid: (K/32, N/32).
// ---------------------------------------------------------------------------
__global__ __launch_bounds__(256) void castT(const float* __restrict__ W,
                                             unsigned short* __restrict__ Wt,
                                             int K, int N) {
  __shared__ float tile[32][33];
  const int k0 = blockIdx.x * 32, n0 = blockIdx.y * 32;
  const int tx = threadIdx.x & 31, ty = threadIdx.x >> 5;  // 32 x 8
#pragma unroll
  for (int i = ty; i < 32; i += 8)
    tile[i][tx] = W[(size_t)(k0 + i) * N + n0 + tx];
  __syncthreads();
#pragma unroll
  for (int i = ty; i < 32; i += 8)
    Wt[(size_t)(n0 + tx) * K + k0 + i] = f2bf(tile[i][tx]);
}

// ---------------------------------------------------------------------------
// LayerNorm fp32 -> bf16.  One block per row.
// ---------------------------------------------------------------------------
__global__ __launch_bounds__(256) void layernorm_bf16(
    const float* __restrict__ x, const float* __restrict__ g,
    const float* __restrict__ bta, unsigned short* __restrict__ out, int C) {
  __shared__ float red[256];
  const int row = blockIdx.x, tid = threadIdx.x;
  const float* xr = x + (size_t)row * C;
  float s = 0.0f;
  for (int i = tid; i < C; i += 256) s += xr[i];
  red[tid] = s; __syncthreads();
  for (int st = 128; st > 0; st >>= 1) {
    if (tid < st) red[tid] += red[tid + st];
    __syncthreads();
  }
  const float mu = red[0] / (float)C;
  __syncthreads();
  float v = 0.0f;
  for (int i = tid; i < C; i += 256) { float d = xr[i] - mu; v += d * d; }
  red[tid] = v; __syncthreads();
  for (int st = 128; st > 0; st >>= 1) {
    if (tid < st) red[tid] += red[tid + st];
    __syncthreads();
  }
  const float rs = rsqrtf(red[0] / (float)C + 1e-5f);
  unsigned short* orow = out + (size_t)row * C;
  for (int i = tid; i < C; i += 256)
    orow[i] = f2bf((xr[i] - mu) * rs * g[i] + bta[i]);
}

// ---------------------------------------------------------------------------
// GEMM: out[M][N] = A[M][K](bf16) * Wt[N][K]^T(bf16) + bias, epilogue modes:
//   mode 0: bf16 out (bias only)          -> outb
//   mode 1: fp32 out = resid + val        -> outf
//   mode 2: bf16 out = gelu(val)          -> outb
// Block tile 64(M) x 256(N), 8 waves of 32x64 each, K step 32.
// Tiles stream via GLOBAL_LOAD_ASYNC_TO_LDS_B128, double buffered so tile t+1
// loads overlap tile t WMMA work.
// ---------------------------------------------------------------------------
__global__ __launch_bounds__(256) void gemm_bf16_wmma(
    const unsigned short* __restrict__ A, const unsigned short* __restrict__ Wt,
    const float* __restrict__ bias, const float* __restrict__ resid,
    float* __restrict__ outf, unsigned short* __restrict__ outb,
    int M, int N, int K, int mode) {
  __shared__ __align__(16) unsigned int lA[2][64 * 16];    //  64 x 32 bf16
  __shared__ __align__(16) unsigned int lB[2][256 * 16];   // 256 x 32 bf16
  const int tid  = threadIdx.x;
  const int m0   = blockIdx.y * 64;
  const int n0   = blockIdx.x * 256;
  const int wid  = tid >> 5, lane = tid & 31;
  const int half = lane >> 4, nl = lane & 15;
  const int wr   = wid & 1, wc = wid >> 1;
  const int Ku   = K >> 1;
  const unsigned int* Au = (const unsigned int*)A;
  const unsigned int* Wu = (const unsigned int*)Wt;

  FragC acc[2][4];
#pragma unroll
  for (int i = 0; i < 2; ++i)
#pragma unroll
    for (int j = 0; j < 4; ++j)
#pragma unroll
      for (int r = 0; r < 8; ++r) acc[i][j].f[r] = 0.0f;

  // Issue one K-step tile (A: 4KB, B: 16KB) as async b128 copies.
  auto issue = [&](int buf, int k0) {
    {
      int i = tid;                       // A tile: 256 b128 transfers
      int row = i >> 2, cp = (i & 3) * 4;
      async_b128(lds_lo(&lA[buf][i * 4]),
                 (const void*)(Au + (size_t)(m0 + row) * Ku + (k0 >> 1) + cp));
    }
#pragma unroll
    for (int t = 0; t < 4; ++t) {        // B tile: 1024 b128 transfers
      int j = tid + t * 256;
      int row = j >> 2, cp = (j & 3) * 4;
      async_b128(lds_lo(&lB[buf][j * 4]),
                 (const void*)(Wu + (size_t)(n0 + row) * Ku + (k0 >> 1) + cp));
    }
  };

  const int ntiles = K >> 5;
  issue(0, 0);
  for (int t = 0; t < ntiles; ++t) {
    const int cur = t & 1;
    wait_async0();                       // my tile-t transfers complete
    __syncthreads();                     // everyone's tile t visible; tile t-1
                                         // compute done by all waves
    if (t + 1 < ntiles) issue(cur ^ 1, (t + 1) * 32);

    Frag16 a[2];
#pragma unroll
    for (int s = 0; s < 2; ++s) {        // A fragments (16x32 layout)
      int mrow = wr * 32 + s * 16 + nl;
#pragma unroll
      for (int v = 0; v < 8; ++v) {
        int kk = 2 * v + half * 16;
        a[s].u[v] = lA[cur][mrow * 16 + (kk >> 1)];
      }
    }
#pragma unroll
    for (int j = 0; j < 4; ++j) {        // B fragments (32x16 layout)
      Frag16 b;
      int ncol = wc * 64 + j * 16 + nl;
#pragma unroll
      for (int v = 0; v < 8; ++v) {
        int kk = 2 * v + half * 16;
        b.u[v] = lB[cur][ncol * 16 + (kk >> 1)];
      }
      acc[0][j].v = __builtin_amdgcn_wmma_f32_16x16x32_bf16(
          false, a[0].v, false, b.v, (short)0, acc[0][j].v, false, false);
      acc[1][j].v = __builtin_amdgcn_wmma_f32_16x16x32_bf16(
          false, a[1].v, false, b.v, (short)0, acc[1][j].v, false, false);
    }
  }

#pragma unroll
  for (int i = 0; i < 2; ++i) {
#pragma unroll
    for (int j = 0; j < 4; ++j) {
      int n = n0 + wc * 64 + j * 16 + nl;
      float bn = bias[n];
#pragma unroll
      for (int r = 0; r < 8; ++r) {
        int m = m0 + wr * 32 + i * 16 + r + half * 8;
        float val = acc[i][j].f[r] + bn;
        size_t idx = (size_t)m * N + n;
        if (mode == 0) {
          outb[idx] = f2bf(val);
        } else if (mode == 1) {
          outf[idx] = resid[idx] + val;
        } else {
          float t2 = val + 0.044715f * val * val * val;
          outb[idx] = f2bf(0.5f * val * (1.0f + tanhf(0.7978845608028654f * t2)));
        }
      }
    }
  }
}

// ---------------------------------------------------------------------------
// Flash attention.  qkv: bf16 [B*T][3C] (q|k|v per head slice of hd=64).
// Grid (T/128, H, B); each of 8 waves owns 16 query rows; KV streamed in
// 32-row tiles through LDS with online softmax in registers.
// ---------------------------------------------------------------------------
__global__ __launch_bounds__(256) void attn_wmma(
    const unsigned short* __restrict__ qkv, unsigned short* __restrict__ y,
    int T, int H, int hd) {
  const int C = H * hd, C3 = 3 * C;
  const int b = blockIdx.z, h = blockIdx.y;
  const int tid = threadIdx.x, wid = tid >> 5, lane = tid & 31;
  const int half = lane >> 4, nl = lane & 15;
  const int q0 = blockIdx.x * 128 + wid * 16;

  __shared__ unsigned int Ktu[32 * 32];      // K tile [kv 32][d 64] bf16
  __shared__ unsigned int Vtu[64 * 16];      // V tile transposed [d 64][kv 32]
  __shared__ unsigned short Ps[8][16 * 32];  // per-wave P staging (16 x 32)
  const unsigned int* qu = (const unsigned int*)qkv;

  // Q fragments: rows q0..q0+15, hd split into two 32-wide K chunks.
  Frag16 qa[2];
#pragma unroll
  for (int s = 0; s < 2; ++s)
#pragma unroll
    for (int v = 0; v < 8; ++v) {
      int kk = 2 * v + half * 16 + s * 32;
      qa[s].u[v] = qu[((size_t)(b * T + q0 + nl) * C3 + h * hd + kk) >> 1];
    }

  FragC o[4];
  float mrow[8], lrow[8];
#pragma unroll
  for (int j = 0; j < 4; ++j)
#pragma unroll
    for (int r = 0; r < 8; ++r) o[j].f[r] = 0.0f;
#pragma unroll
  for (int r = 0; r < 8; ++r) { mrow[r] = -3.4e38f; lrow[r] = 0.0f; }

  const int kvend = blockIdx.x * 128 + 128;  // causal bound for whole block
  for (int kv0 = 0; kv0 < kvend; kv0 += 32) {
    __syncthreads();
#pragma unroll
    for (int it = 0; it < 4; ++it) {         // K tile (row-contiguous)
      int i = tid + it * 256;
      int row = i >> 5, cp = i & 31;
      Ktu[i] = qu[(((size_t)(b * T + kv0 + row) * C3 + C + h * hd) >> 1) + cp];
    }
#pragma unroll
    for (int it = 0; it < 8; ++it) {         // V tile (transposed gather)
      int i = tid + it * 256;
      int d = i >> 5, kv = i & 31;
      ((unsigned short*)Vtu)[d * 32 + kv] =
          qkv[(size_t)(b * T + kv0 + kv) * C3 + 2 * C + h * hd + d];
    }
    __syncthreads();
    if (kv0 > q0 + 15) continue;             // wave-uniform; EXEC stays full

    // S = Q * K^T : two 16-col halves, hd=64 => two K-steps each.
    Frag16 kb;
    FragC s0, s1;
#pragma unroll
    for (int r = 0; r < 8; ++r) { s0.f[r] = 0.0f; s1.f[r] = 0.0f; }
#pragma unroll
    for (int ks = 0; ks < 2; ++ks) {
#pragma unroll
      for (int v = 0; v < 8; ++v) {
        int kk = 2 * v + half * 16 + ks * 32;
        kb.u[v] = Ktu[nl * 32 + (kk >> 1)];
      }
      s0.v = __builtin_amdgcn_wmma_f32_16x16x32_bf16(
          false, qa[ks].v, false, kb.v, (short)0, s0.v, false, false);
#pragma unroll
      for (int v = 0; v < 8; ++v) {
        int kk = 2 * v + half * 16 + ks * 32;
        kb.u[v] = Ktu[(16 + nl) * 32 + (kk >> 1)];
      }
      s1.v = __builtin_amdgcn_wmma_f32_16x16x32_bf16(
          false, qa[ks].v, false, kb.v, (short)0, s1.v, false, false);
    }

    // Online softmax update (C layout: row = r + 8*half, col = nl [+16]).
    float p0a[8], p1a[8];
#pragma unroll
    for (int r = 0; r < 8; ++r) {
      int rowg = q0 + r + half * 8;
      float v0 = s0.f[r] * 0.125f;           // 1/sqrt(64)
      float v1 = s1.f[r] * 0.125f;
      if (kv0 + nl > rowg)      v0 = -3.4e38f;
      if (kv0 + 16 + nl > rowg) v1 = -3.4e38f;
      float mn = fmaxf(mrow[r], redmax16(fmaxf(v0, v1)));
      float alpha = __expf(mrow[r] - mn);
      mrow[r] = mn;
      float p0 = __expf(v0 - mn);
      float p1 = __expf(v1 - mn);
      lrow[r] = lrow[r] * alpha + redsum16(p0 + p1);
      p0a[r] = p0; p1a[r] = p1;
#pragma unroll
      for (int j = 0; j < 4; ++j) o[j].f[r] *= alpha;
    }

    // Stage P through LDS to reshape C-layout -> A-layout.
    unsigned short* pw = Ps[wid];
#pragma unroll
    for (int r = 0; r < 8; ++r) {
      pw[(r + half * 8) * 32 + nl]      = f2bf(p0a[r]);
      pw[(r + half * 8) * 32 + nl + 16] = f2bf(p1a[r]);
    }
    asm volatile("s_wait_dscnt 0x0" ::: "memory");
    __builtin_amdgcn_wave_barrier();

    Frag16 pa;
    const unsigned int* pu = (const unsigned int*)pw;
#pragma unroll
    for (int v = 0; v < 8; ++v) {
      int kk = 2 * v + half * 16;
      pa.u[v] = pu[nl * 16 + (kk >> 1)];
    }
#pragma unroll
    for (int j = 0; j < 4; ++j) {            // O += P * V (4 d-subtiles)
      Frag16 vb;
#pragma unroll
      for (int v = 0; v < 8; ++v) {
        int kk = 2 * v + half * 16;
        vb.u[v] = Vtu[(j * 16 + nl) * 16 + (kk >> 1)];
      }
      o[j].v = __builtin_amdgcn_wmma_f32_16x16x32_bf16(
          false, pa.v, false, vb.v, (short)0, o[j].v, false, false);
    }
  }

#pragma unroll
  for (int j = 0; j < 4; ++j)
#pragma unroll
    for (int r = 0; r < 8; ++r) {
      int rowg = q0 + r + half * 8;
      int d = j * 16 + nl;
      y[(size_t)(b * T + rowg) * C + h * hd + d] = f2bf(o[j].f[r] / lrow[r]);
    }
}

// ---------------------------------------------------------------------------
// Launcher
// ---------------------------------------------------------------------------
extern "C" void kernel_launch(void* const* d_in, const int* in_sizes, int n_in,
                              void* d_out, int out_size, void* d_ws, size_t ws_size,
                              hipStream_t stream) {
  (void)in_sizes; (void)n_in; (void)out_size; (void)ws_size;
  const int Bz = 2, T = 2048, C = 768, H = 12, hd = 64;
  const int M = Bz * T;        // 4096
  const int C3 = 3 * C;        // 2304
  const int C4 = 4 * C;        // 3072

  const float* x      = (const float*)d_in[0];
  const float* ln1_g  = (const float*)d_in[1];
  const float* ln1_b  = (const float*)d_in[2];
  const float* w_attn = (const float*)d_in[3];
  const float* b_attn = (const float*)d_in[4];
  const float* w_proj = (const float*)d_in[5];
  const float* b_proj = (const float*)d_in[6];
  const float* ln2_g  = (const float*)d_in[7];
  const float* ln2_b  = (const float*)d_in[8];
  const float* w_fc   = (const float*)d_in[9];
  const float* b_fc   = (const float*)d_in[10];
  const float* w_mlp  = (const float*)d_in[11];
  const float* b_mlp  = (const float*)d_in[12];
  float* out = (float*)d_out;

  char* ws = (char*)d_ws;
  size_t off = 0;
  auto alloc = [&](size_t bytes) -> void* {
    void* p = ws + off;
    off = (off + bytes + 255) & ~(size_t)255;
    return p;
  };
  unsigned short* wtA = (unsigned short*)alloc((size_t)C3 * C * 2);
  unsigned short* wtP = (unsigned short*)alloc((size_t)C * C * 2);
  unsigned short* wtF = (unsigned short*)alloc((size_t)C4 * C * 2);
  unsigned short* wtM = (unsigned short*)alloc((size_t)C * C4 * 2);
  unsigned short* h1  = (unsigned short*)alloc((size_t)M * C * 2);
  unsigned short* qkv = (unsigned short*)alloc((size_t)M * C3 * 2);
  unsigned short* yb  = (unsigned short*)alloc((size_t)M * C * 2);
  float*          x2  = (float*)alloc((size_t)M * C * 4);
  unsigned short* h2  = (unsigned short*)alloc((size_t)M * C * 2);
  unsigned short* fca = (unsigned short*)alloc((size_t)M * C4 * 2);

  dim3 blk(256);
  castT<<<dim3(C / 32, C3 / 32), blk, 0, stream>>>(w_attn, wtA, C, C3);
  castT<<<dim3(C / 32, C / 32),  blk, 0, stream>>>(w_proj, wtP, C, C);
  castT<<<dim3(C / 32, C4 / 32), blk, 0, stream>>>(w_fc,   wtF, C, C4);
  castT<<<dim3(C4 / 32, C / 32), blk, 0, stream>>>(w_mlp,  wtM, C4, C);

  layernorm_bf16<<<M, blk, 0, stream>>>(x, ln1_g, ln1_b, h1, C);
  gemm_bf16_wmma<<<dim3(C3 / 256, M / 64), blk, 0, stream>>>(
      h1, wtA, b_attn, nullptr, nullptr, qkv, M, C3, C, 0);
  attn_wmma<<<dim3(T / 128, H, Bz), blk, 0, stream>>>(qkv, yb, T, H, hd);
  gemm_bf16_wmma<<<dim3(C / 256, M / 64), blk, 0, stream>>>(
      yb, wtP, b_proj, x, x2, nullptr, M, C, C, 1);
  layernorm_bf16<<<M, blk, 0, stream>>>(x2, ln2_g, ln2_b, h2, C);
  gemm_bf16_wmma<<<dim3(C4 / 256, M / 64), blk, 0, stream>>>(
      h2, wtF, b_fc, nullptr, nullptr, fca, M, C4, C, 2);
  gemm_bf16_wmma<<<dim3(C / 256, M / 64), blk, 0, stream>>>(
      fca, wtM, b_mlp, x2, out, nullptr, M, C, C4, 1);
}
